// CrossWinAttentionNystrom_29583734735186
// MI455X (gfx1250) — compile-verified, matrix operand order checked by reading, not run on hardware
//
#include <hip/hip_runtime.h>

// ---------------------------------------------------------------------------
// CrossWinAttentionNystrom fused kernel for gfx1250 (MI455X, wave32, WMMA).
// One workgroup per (b, l) location. bf16 WMMA everywhere; Newton-Schulz
// pinv runs one-wave-per-head (barrier-free, LDS in-order within a wave),
// with transposed B-operand layouts so all NS LDS traffic is b128 packed.
// ---------------------------------------------------------------------------

typedef __attribute__((ext_vector_type(16))) __bf16 v16bf;
typedef __attribute__((ext_vector_type(8)))  float  v8f;
typedef __attribute__((ext_vector_type(2)))  __bf16 bf16x2;
typedef __attribute__((ext_vector_type(8)))  __bf16 bf16x8;

#define N_HEADS   4
#define DH        32
#define DIMP      128
#define SK        96
#define ML        32
#define NS_ITERS  6
#define LN_EPS    1e-5f
#define SM_EPS    1e-6f
#define DAMP_EPS  1e-4f
#define Q_SCALE   0.17677669529663687f   // 32^-0.5

// landmark indices: unique(rint(linspace(0, 95, 32)))
__device__ __constant__ int LAND_IDX[ML] = {
  0,3,6,9,12,15,18,21,25,28,31,34,37,40,43,46,
  49,52,55,58,61,64,67,70,74,77,80,83,86,89,92,95};

__device__ __forceinline__ float wsum(float v) {
  #pragma unroll
  for (int m = 16; m; m >>= 1) v += __shfl_xor(v, m, 32);
  return v;
}
__device__ __forceinline__ float wmaxr(float v) {
  #pragma unroll
  for (int m = 16; m; m >>= 1) v = fmaxf(v, __shfl_xor(v, m, 32));
  return v;
}

#define POST_PLAIN 0
#define POST_7I    1
#define POST_15I   2
#define POST_13I   3
#define POST_QTR   4

__device__ __forceinline__ float ns_post(int mode, float v, bool diag) {
  switch (mode) {
    case POST_7I:  return (diag ? 7.f  : 0.f) - v;
    case POST_15I: return (diag ? 15.f : 0.f) - v;
    case POST_13I: return (diag ? 13.f : 0.f) - v;
    case POST_QTR: return 0.25f * v;
    default:       return v;
  }
}

// Single-wave 32x32x32 bf16 matmul: acc = X @ B where B[k][n] = Yt[n*32+k]
// (Yt is the transposed-stored right operand => packed b128 fragment loads).
// Optional outputs: outR row-major (strided b16 stores), outT transposed
// (single packed 16B store per tile). All loads precede all stores, so
// in-place updates of X/Yt buffers are safe within the wave.
__device__ __forceinline__ void mm32_1w(const __bf16* X, const __bf16* Yt,
                                        __bf16* outR, int postR,
                                        __bf16* outT, int postT,
                                        int lane) {
  const int lm = lane & 15, kh2 = lane >> 4;
  v16bf afr[2], bfr[2];
  #pragma unroll
  for (int t = 0; t < 2; ++t) {
    const uint* ar = (const uint*)(X  + (t * 16 + lm) * 32);
    const uint* br = (const uint*)(Yt + (t * 16 + lm) * 32);
    #pragma unroll
    for (int i = 0; i < 8; ++i) {
      const int kb = ((i & 3) << 1) + ((i >> 2) << 4) + (kh2 << 3);
      bf16x2 pa = __builtin_bit_cast(bf16x2, ar[kb >> 1]);
      bf16x2 pb = __builtin_bit_cast(bf16x2, br[kb >> 1]);
      afr[t][2 * i] = pa.x; afr[t][2 * i + 1] = pa.y;
      bfr[t][2 * i] = pb.x; bfr[t][2 * i + 1] = pb.y;
    }
  }
  v8f acc[2][2];
  #pragma unroll
  for (int mt = 0; mt < 2; ++mt)
    #pragma unroll
    for (int nt = 0; nt < 2; ++nt) {
      v8f c = {};
      acc[mt][nt] = __builtin_amdgcn_wmma_f32_16x16x32_bf16(
          false, afr[mt], false, bfr[nt], (short)0, c, false, false);
    }
  #pragma unroll
  for (int mt = 0; mt < 2; ++mt)
    #pragma unroll
    for (int nt = 0; nt < 2; ++nt) {
      const int N = nt * 16 + lm;
      if (outT) {
        bf16x8 vv;
        #pragma unroll
        for (int r = 0; r < 8; ++r) {
          const int M = mt * 16 + kh2 * 8 + r;
          vv[r] = (__bf16)ns_post(postT, acc[mt][nt][r], M == N);
        }
        *(bf16x8*)(outT + N * 32 + mt * 16 + kh2 * 8) = vv;
      }
      if (outR) {
        #pragma unroll
        for (int r = 0; r < 8; ++r) {
          const int M = mt * 16 + kh2 * 8 + r;
          outR[M * 32 + N] = (__bf16)ns_post(postR, acc[mt][nt][r], M == N);
        }
      }
    }
}

// prep: transpose w_k / w_v (128x128 f32, [d][o]) to bf16 [o][d] in workspace
__global__ void nystrom_prep_weights(const float* __restrict__ wk,
                                     const float* __restrict__ wv,
                                     __bf16* __restrict__ wkT,
                                     __bf16* __restrict__ wvT) {
  const int i = blockIdx.x * blockDim.x + threadIdx.x;  // 0..16383
  const int o = i >> 7, d = i & 127;
  wkT[o * 128 + d] = (__bf16)wk[d * 128 + o];
  wvT[o * 128 + d] = (__bf16)wv[d * 128 + o];
}

__global__ __launch_bounds__(256)
void nystrom_attn_kernel(const float* __restrict__ qg,
                         const float* __restrict__ kg,
                         const float* __restrict__ vg,
                         const float* __restrict__ ln_q_w, const float* __restrict__ ln_q_b,
                         const float* __restrict__ ln_k_w, const float* __restrict__ ln_k_b,
                         const float* __restrict__ ln_v_w, const float* __restrict__ ln_v_b,
                         const float* __restrict__ w_q,    const float* __restrict__ b_q,
                         const float* __restrict__ b_k,    const float* __restrict__ b_v,
                         const float* __restrict__ w_proj, const float* __restrict__ b_proj,
                         const __bf16* __restrict__ wkT,   const __bf16* __restrict__ wvT,
                         float* __restrict__ outg) {
  extern __shared__ __align__(16) char smem[];
  __bf16* nbuf = (__bf16*)smem;            // 96x128: normed K/V, then projected V
  __bf16* kpb  = nbuf + SK * DIMP;         // 96x128: projected K
  __bf16* kl4  = kpb + SK * DIMP;          // 4 x 32x32 landmark tiles
  __bf16* Pf4  = kl4 + 4 * 1024;           // 4 x 32x96 P (raw -> softmaxed)
  __bf16* NS   = Pf4 + 4 * 3072;           // 4 heads x {Ah,Zr,Zt,Ta,Tbt,Tct}
  float*  Sf4  = (float*)(NS + 4 * 6144);  // 4 x 32x32 raw landmark Gram
  float*  qn   = Sf4 + 4096;               // 128 LN'd q
  float*  qps  = qn + 128;                 // 128 projected+scaled q
  float*  kql4 = qps + 128;                // 4 x 32
  float*  r14  = kql4 + 128;               // 4 x 32
  float*  r24  = r14 + 128;                // 4 x 96
  float*  outv = r24 + 384;                // 128 attn output (pre-proj)

  const int tid  = threadIdx.x;
  const int lane = tid & 31, wid = tid >> 5;
  const int bl   = blockIdx.x;             // 0..2047
  const int bb   = bl >> 10, l = bl & 1023;
  const int d0   = lane * 4;
  const int lm   = lane & 15, kh2 = lane >> 4;

  // ---------------- Phase 1: load + LN K rows -> nbuf (bf16) ----------------
  {
    float lw[4], lb[4];
    #pragma unroll
    for (int j = 0; j < 4; ++j) { lw[j] = ln_k_w[d0 + j]; lb[j] = ln_k_b[d0 + j]; }
    for (int rr = 0; rr < 12; ++rr) {
      const int s = wid * 12 + rr;
      const int n = s >> 4, ij = s & 15;
      const size_t base = ((size_t)((bb * 6 + n) * 1024 + l)) * 2048 + (size_t)ij * 128;
      float4 x4 = *(const float4*)(kg + base + d0);
      __builtin_prefetch(vg + base + d0, 0, 1);   // global_prefetch_b8 for V tile
      float xa[4] = {x4.x, x4.y, x4.z, x4.w};
      float mu = wsum(xa[0] + xa[1] + xa[2] + xa[3]) * (1.f / 128.f);
      float sq = 0.f;
      #pragma unroll
      for (int j = 0; j < 4; ++j) { float dd = xa[j] - mu; sq += dd * dd; }
      float rstd = rsqrtf(wsum(sq) * (1.f / 128.f) + LN_EPS);
      #pragma unroll
      for (int j = 0; j < 4; ++j)
        nbuf[s * DIMP + d0 + j] = (__bf16)((xa[j] - mu) * rstd * lw[j] + lb[j]);
    }
    if (wid == 0) {  // q row LN
      const size_t qb = ((size_t)(bb * 1024 + l)) * 128;
      float4 x4 = *(const float4*)(qg + qb + d0);
      float xa[4] = {x4.x, x4.y, x4.z, x4.w};
      float mu = wsum(xa[0] + xa[1] + xa[2] + xa[3]) * (1.f / 128.f);
      float sq = 0.f;
      #pragma unroll
      for (int j = 0; j < 4; ++j) { float dd = xa[j] - mu; sq += dd * dd; }
      float rstd = rsqrtf(wsum(sq) * (1.f / 128.f) + LN_EPS);
      #pragma unroll
      for (int j = 0; j < 4; ++j)
        qn[d0 + j] = (xa[j] - mu) * rstd * ln_q_w[d0 + j] + ln_q_b[d0 + j];
    }
  }
  __syncthreads();

  // ---------------- Phase 2: Kp = LN(K) @ w_k + b_k (WMMA) -> kpb -----------
  {
    const int nt = wid;                        // 16-col strip per wave
    const float bk = b_k[nt * 16 + lm];
    for (int mt = 0; mt < 6; ++mt) {
      v8f c;
      #pragma unroll
      for (int r = 0; r < 8; ++r) c[r] = bk;
      const uint* arow = (const uint*)(nbuf + (mt * 16 + lm) * DIMP);
      const uint* brow = (const uint*)(wkT + (nt * 16 + lm) * DIMP);
      #pragma unroll
      for (int ks = 0; ks < 4; ++ks) {
        v16bf a, b;
        #pragma unroll
        for (int i = 0; i < 8; ++i) {
          const int kb = ks * 32 + ((i & 3) << 1) + ((i >> 2) << 4) + (kh2 << 3);
          bf16x2 pa = __builtin_bit_cast(bf16x2, arow[kb >> 1]);
          bf16x2 pb = __builtin_bit_cast(bf16x2, brow[kb >> 1]);
          a[2 * i] = pa.x; a[2 * i + 1] = pa.y;
          b[2 * i] = pb.x; b[2 * i + 1] = pb.y;
        }
        c = __builtin_amdgcn_wmma_f32_16x16x32_bf16(false, a, false, b,
                                                    (short)0, c, false, false);
      }
      #pragma unroll
      for (int r = 0; r < 8; ++r)
        kpb[(mt * 16 + r + kh2 * 8) * DIMP + nt * 16 + lm] = (__bf16)c[r];
    }
  }
  __syncthreads();

  // ---------------- Phase 3: load + LN V rows -> nbuf; q projection ---------
  {
    float lw[4], lb[4];
    #pragma unroll
    for (int j = 0; j < 4; ++j) { lw[j] = ln_v_w[d0 + j]; lb[j] = ln_v_b[d0 + j]; }
    for (int rr = 0; rr < 12; ++rr) {
      const int s = wid * 12 + rr;
      const int n = s >> 4, ij = s & 15;
      const size_t base = ((size_t)((bb * 6 + n) * 1024 + l)) * 2048 + (size_t)ij * 128;
      float4 x4 = *(const float4*)(vg + base + d0);
      float xa[4] = {x4.x, x4.y, x4.z, x4.w};
      float mu = wsum(xa[0] + xa[1] + xa[2] + xa[3]) * (1.f / 128.f);
      float sq = 0.f;
      #pragma unroll
      for (int j = 0; j < 4; ++j) { float dd = xa[j] - mu; sq += dd * dd; }
      float rstd = rsqrtf(wsum(sq) * (1.f / 128.f) + LN_EPS);
      #pragma unroll
      for (int j = 0; j < 4; ++j)
        nbuf[s * DIMP + d0 + j] = (__bf16)((xa[j] - mu) * rstd * lw[j] + lb[j]);
    }
    if (tid < 128) {   // qps = (LN(q) @ w_q + b_q) * scale
      float acc = b_q[tid];
      for (int dd = 0; dd < 128; ++dd) acc += qn[dd] * w_q[dd * 128 + tid];
      qps[tid] = acc * Q_SCALE;
    }
  }
  __syncthreads();

  // ---------------- Phase 4: Vp = LN(V) @ w_v + b_v (WMMA) -> nbuf ----------
  {
    const int nt = wid;
    const float bv = b_v[nt * 16 + lm];
    v8f accv[6];
    for (int mt = 0; mt < 6; ++mt) {
      v8f c;
      #pragma unroll
      for (int r = 0; r < 8; ++r) c[r] = bv;
      const uint* arow = (const uint*)(nbuf + (mt * 16 + lm) * DIMP);
      const uint* brow = (const uint*)(wvT + (nt * 16 + lm) * DIMP);
      #pragma unroll
      for (int ks = 0; ks < 4; ++ks) {
        v16bf a, b;
        #pragma unroll
        for (int i = 0; i < 8; ++i) {
          const int kb = ks * 32 + ((i & 3) << 1) + ((i >> 2) << 4) + (kh2 << 3);
          bf16x2 pa = __builtin_bit_cast(bf16x2, arow[kb >> 1]);
          bf16x2 pb = __builtin_bit_cast(bf16x2, brow[kb >> 1]);
          a[2 * i] = pa.x; a[2 * i + 1] = pa.y;
          b[2 * i] = pb.x; b[2 * i + 1] = pb.y;
        }
        c = __builtin_amdgcn_wmma_f32_16x16x32_bf16(false, a, false, b,
                                                    (short)0, c, false, false);
      }
      accv[mt] = c;
    }
    __syncthreads();   // all reads of nbuf (normed V) complete
    for (int mt = 0; mt < 6; ++mt) {
      #pragma unroll
      for (int r = 0; r < 8; ++r)
        nbuf[(mt * 16 + r + kh2 * 8) * DIMP + nt * 16 + lm] = (__bf16)accv[mt][r];
    }
  }
  __syncthreads();

  // ---------------- Phase 5a: gather landmark tiles for all heads -----------
  for (int e = tid; e < N_HEADS * ML * DH; e += 256) {
    const int head = e >> 10, i = (e >> 5) & 31, dd = e & 31;
    kl4[e] = kpb[LAND_IDX[i] * DIMP + head * DH + dd];
  }
  __syncthreads();

  // ---------------- Phase 5b: S (16 tiles) + P (48 tiles), all heads --------
  for (int rep = 0; rep < 8; ++rep) {
    const int t = wid * 8 + rep;
    v16bf a, b;
    v8f c = {};
    if (t < 16) {                 // S = kl @ kl^T for head t>>2
      const int head = t >> 2, mt = (t >> 1) & 1, nt = t & 1;
      const __bf16* klh = kl4 + head * 1024;
      const uint* arow = (const uint*)(klh + (mt * 16 + lm) * 32);
      const uint* brow = (const uint*)(klh + (nt * 16 + lm) * 32);
      #pragma unroll
      for (int i = 0; i < 8; ++i) {
        const int kb = ((i & 3) << 1) + ((i >> 2) << 4) + (kh2 << 3);
        bf16x2 pa = __builtin_bit_cast(bf16x2, arow[kb >> 1]);
        bf16x2 pb = __builtin_bit_cast(bf16x2, brow[kb >> 1]);
        a[2 * i] = pa.x; a[2 * i + 1] = pa.y;
        b[2 * i] = pb.x; b[2 * i + 1] = pb.y;
      }
      c = __builtin_amdgcn_wmma_f32_16x16x32_bf16(false, a, false, b,
                                                  (short)0, c, false, false);
      #pragma unroll
      for (int r = 0; r < 8; ++r)
        Sf4[head * 1024 + (mt * 16 + r + kh2 * 8) * 32 + nt * 16 + lm] = c[r];
    } else {                      // P = kl @ kh^T
      const int u = t - 16;
      const int head = u / 12, rem = u % 12;
      const int mt = rem & 1, nt6 = rem >> 1;
      const __bf16* klh = kl4 + head * 1024;
      const uint* arow = (const uint*)(klh + (mt * 16 + lm) * 32);
      const uint* brow = (const uint*)(kpb + (nt6 * 16 + lm) * DIMP + head * DH);
      #pragma unroll
      for (int i = 0; i < 8; ++i) {
        const int kb = ((i & 3) << 1) + ((i >> 2) << 4) + (kh2 << 3);
        bf16x2 pa = __builtin_bit_cast(bf16x2, arow[kb >> 1]);
        bf16x2 pb = __builtin_bit_cast(bf16x2, brow[kb >> 1]);
        a[2 * i] = pa.x; a[2 * i + 1] = pa.y;
        b[2 * i] = pb.x; b[2 * i + 1] = pb.y;
      }
      c = __builtin_amdgcn_wmma_f32_16x16x32_bf16(false, a, false, b,
                                                  (short)0, c, false, false);
      #pragma unroll
      for (int r = 0; r < 8; ++r)
        Pf4[head * 3072 + (mt * 16 + r + kh2 * 8) * 96 + nt6 * 16 + lm] = (__bf16)c[r];
    }
  }
  __syncthreads();

  // ---- Phase 5c: waves 0-3: per-head softmax(A), K_qL, Z0, Newton-Schulz,
  //                r1 — entirely wave-local (LDS in-order within a wave).
  //      waves 4-7: per-head P row softmax (runs concurrently).
  if (wid < 4) {
    const int h = wid;
    const __bf16* klh = kl4 + h * 1024;
    __bf16* Ah  = NS + h * 6144;
    __bf16* Zr  = Ah + 1024;
    __bf16* Zt  = Ah + 2048;
    __bf16* Ta  = Ah + 3072;
    __bf16* Tbt = Ah + 4096;
    __bf16* Tct = Ah + 5120;
    const float* Sfh = Sf4 + h * 1024;

    // softmax(A) rows + damping
    {
      float mx = -3.4e38f;
      for (int cc = 0; cc < 32; ++cc) mx = fmaxf(mx, Sfh[lane * 32 + cc]);
      float s = 0.f;
      for (int cc = 0; cc < 32; ++cc) s += expf(Sfh[lane * 32 + cc] - mx);
      const float inv = 1.f / (s + SM_EPS);
      for (int cc = 0; cc < 32; ++cc) {
        float av = expf(Sfh[lane * 32 + cc] - mx) * inv;
        if (cc == lane) av += DAMP_EPS;
        Ah[lane * 32 + cc] = (__bf16)av;
      }
    }
    __builtin_amdgcn_wave_barrier();

    // K_qL = softmax(q_h . k_land)
    {
      float dq = 0.f;
      for (int dd = 0; dd < 32; ++dd)
        dq += qps[h * DH + dd] * (float)klh[lane * 32 + dd];
      const float mx = wmaxr(dq);
      const float e  = expf(dq - mx);
      const float s  = wsum(e);
      kql4[h * 32 + lane] = e / (s + SM_EPS);
    }

    // Z0 = A^T / (||A||_1 * ||A||_inf); Zt = transposed copy (== scaled A)
    {
      float cs = 0.f, rs = 0.f;
      for (int r = 0; r < 32; ++r) cs += (float)Ah[r * 32 + lane];
      for (int cc = 0; cc < 32; ++cc) rs += (float)Ah[lane * 32 + cc];
      const float al = 1.f / (wmaxr(cs) * wmaxr(rs));
      for (int r = 0; r < 32; ++r)
        Zr[lane * 32 + r] = (__bf16)((float)Ah[r * 32 + lane] * al);
      for (int cc = 0; cc < 32; ++cc)
        Zt[lane * 32 + cc] = (__bf16)((float)Ah[lane * 32 + cc] * al);
    }
    __builtin_amdgcn_wave_barrier();

    // Newton-Schulz: Z <- 0.25 Z (13I - AZ(15I - AZ(7I - AZ)))
    for (int it = 0; it < NS_ITERS; ++it) {
      mm32_1w(Ah, Zt, Ta, POST_PLAIN, Tbt, POST_7I, lane);   // Ta=AZ, Tbt=(7I-AZ)^T
      __builtin_amdgcn_wave_barrier();
      mm32_1w(Ta, Tbt, nullptr, 0, Tct, POST_15I, lane);     // Tct=(15I-..)^T
      __builtin_amdgcn_wave_barrier();
      mm32_1w(Ta, Tct, nullptr, 0, Tbt, POST_13I, lane);     // Tbt=(13I-..)^T
      __builtin_amdgcn_wave_barrier();
      mm32_1w(Zr, Tbt, Zr, POST_QTR, Zt, POST_QTR, lane);    // Z in place + Z^T
      __builtin_amdgcn_wave_barrier();
    }

    // r1 = K_qL @ pinv(A)   (read via Z^T: contiguous per lane)
    {
      float acc = 0.f;
      for (int m = 0; m < 32; ++m)
        acc += kql4[h * 32 + m] * (float)Zt[lane * 32 + m];
      r14[h * 32 + lane] = acc;
    }
  } else {
    const int h = wid - 4;
    __bf16* Pfh = Pf4 + h * 3072;
    float mx = -3.4e38f;
    for (int cc = 0; cc < 96; ++cc) mx = fmaxf(mx, (float)Pfh[lane * 96 + cc]);
    float s = 0.f;
    for (int cc = 0; cc < 96; ++cc) s += expf((float)Pfh[lane * 96 + cc] - mx);
    const float inv = 1.f / (s + SM_EPS);
    for (int cc = 0; cc < 96; ++cc)
      Pfh[lane * 96 + cc] = (__bf16)(expf((float)Pfh[lane * 96 + cc] - mx) * inv);
  }
  __syncthreads();

  // ---------------- Phase 5d: r2 = r1 @ P (all heads) -----------------------
  for (int e = tid; e < N_HEADS * SK; e += 256) {
    const int hh = e / SK, cc = e % SK;
    float acc = 0.f;
    for (int m = 0; m < 32; ++m)
      acc += r14[hh * 32 + m] * (float)Pf4[hh * 3072 + m * 96 + cc];
    r24[e] = acc;
  }
  __syncthreads();

  // ---------------- Phase 5e: out_h = r2 @ vh -------------------------------
  if (tid < 128) {
    const int hh = tid >> 5, dd = lane;
    float acc = 0.f;
    for (int s = 0; s < SK; ++s)
      acc += r24[hh * 96 + s] * (float)nbuf[s * DIMP + hh * DH + dd];
    outv[tid] = acc;
  }
  __syncthreads();

  // ---------------- Phase 6: output projection ------------------------------
  if (tid < 128) {
    float acc = b_proj[tid];
    for (int hh = 0; hh < 128; ++hh) acc += outv[hh] * w_proj[hh * 128 + tid];
    outg[((size_t)(bb * 1024 + l)) * 128 + tid] = acc;
  }
}

#define SMEM_BYTES 151552

extern "C" void kernel_launch(void* const* d_in, const int* in_sizes, int n_in,
                              void* d_out, int out_size, void* d_ws, size_t ws_size,
                              hipStream_t stream) {
  (void)in_sizes; (void)n_in; (void)out_size; (void)ws_size;
  const float* q       = (const float*)d_in[0];
  const float* k       = (const float*)d_in[1];
  const float* v       = (const float*)d_in[2];
  const float* ln_q_w  = (const float*)d_in[3];
  const float* ln_q_b  = (const float*)d_in[4];
  const float* ln_k_w  = (const float*)d_in[5];
  const float* ln_k_b  = (const float*)d_in[6];
  const float* ln_v_w  = (const float*)d_in[7];
  const float* ln_v_b  = (const float*)d_in[8];
  const float* w_q     = (const float*)d_in[9];
  const float* b_q     = (const float*)d_in[10];
  const float* w_k     = (const float*)d_in[11];
  const float* b_k     = (const float*)d_in[12];
  const float* w_v     = (const float*)d_in[13];
  const float* b_v     = (const float*)d_in[14];
  const float* w_proj  = (const float*)d_in[15];
  const float* b_proj  = (const float*)d_in[16];
  float* out = (float*)d_out;

  __bf16* wkT = (__bf16*)d_ws;
  __bf16* wvT = wkT + 128 * 128;

  nystrom_prep_weights<<<64, 256, 0, stream>>>(w_k, w_v, wkT, wvT);

  hipFuncSetAttribute(reinterpret_cast<const void*>(nystrom_attn_kernel),
                      hipFuncAttributeMaxDynamicSharedMemorySize, SMEM_BYTES);
  nystrom_attn_kernel<<<2048, 256, SMEM_BYTES, stream>>>(
      q, k, v, ln_q_w, ln_q_b, ln_k_w, ln_k_b, ln_v_w, ln_v_b,
      w_q, b_q, b_k, b_v, w_proj, b_proj, wkT, wvT, out);
}
